// MIMO_51934744543718
// MI455X (gfx1250) — matrix operand order repaired and checked by reading, not compiled, not run on hardware
//
#include <hip/hip_runtime.h>
#include <hip/hip_bf16.h>
#include <math.h>

// ---------------------------------------------------------------------------
// Problem constants (from reference)
// ---------------------------------------------------------------------------
#define B_    64
#define T_    512
#define BT    (B_ * T_)      // 32768
#define KIN   32
#define H_    1024
#define H3    3072
#define KOUT  32

// ---------------------------------------------------------------------------
// CDNA5 WMMA types (wave32, 16x16x32 bf16 -> f32)
// ---------------------------------------------------------------------------
typedef __attribute__((ext_vector_type(16))) __bf16      v16bf;
typedef __attribute__((ext_vector_type(8)))  float       v8f;
typedef __attribute__((ext_vector_type(4)))  unsigned int u32x4;
typedef __attribute__((ext_vector_type(4)))  float       f32x4;
// TDM descriptor groups (this toolchain: 6-arg builtin signature
//  (uint32x4 g0, int32x8 g1, int32x4, int32x4, int32x8, i32 cpol))
typedef __attribute__((ext_vector_type(4)))  unsigned int tdm_u4;
typedef __attribute__((ext_vector_type(8)))  int          tdm_i8;
typedef __attribute__((ext_vector_type(4)))  int          tdm_i4;

union Frag32B { v16bf v; u32x4 u[2]; };   // 32 bytes = 16 bf16 = one wmma A/B fragment per lane

static __device__ __forceinline__ v8f wmma_bf16(v16bf a, v16bf b, v8f c) {
    return __builtin_amdgcn_wmma_f32_16x16x32_bf16(
        /*neg_a=*/false, a, /*neg_b=*/false, b,
        /*c_mod=*/(short)0, c, /*reuse_a=*/false, /*reuse_b=*/false);
}

// ---------------------------------------------------------------------------
// Tensor Data Mover: DMA a 2D tile (128 rows x 64B) from a row-major bf16
// matrix with row pitch H_*2 bytes into LDS (contiguous rows of 64B).
// D# bit layout per CDNA5 ISA:
//   group0: [1:0]=count=1, [63:32]=lds_addr, [120:64]=global_addr, [127:126]=2
//   group1: [17:16]=data_size(3 => 8B units), [79:48]=tensor_dim0,
//           [111:80]=tensor_dim1, [127:112]=tile_dim0, [143:128]=tile_dim1,
//           [207:160]=tensor_dim0_stride
// tile_dim0 = 8 units (64B row), tile_dim1 = 128 rows, stride = 256 units.
// ---------------------------------------------------------------------------
static __device__ __forceinline__ void tdm_load_tile(unsigned lds_addr,
                                                     const __bf16* gptr,
                                                     unsigned tensor_rows) {
    const unsigned long long ga = (unsigned long long)(uintptr_t)gptr;
    const unsigned td0   = (H_ * 2) / 8;   // tensor row length in 8B units (256)
    const unsigned td1   = tensor_rows;    // tensor height (rows) for OOB check
    const unsigned tile0 = 8;              // 64B per tile row
    const unsigned tile1 = 128;            // rows per tile

    tdm_u4 g0;
    g0.x = 1u;                                                  // count=1 (valid), user mode
    g0.y = lds_addr;                                            // LDS byte address
    g0.z = (unsigned)ga;                                        // global_addr[31:0]
    g0.w = ((unsigned)(ga >> 32) & 0x01ffffffu) | (2u << 30);   // addr[56:32] | type=2

    tdm_i8 g1;
    g1[0] = (int)(3u << 16);                                    // data_size = 8B units
    g1[1] = (int)((td0 & 0xffffu) << 16);                       // tensor_dim0[15:0]
    g1[2] = (int)((td0 >> 16) | ((td1 & 0xffffu) << 16));       // td0[31:16] | td1[15:0]
    g1[3] = (int)((td1 >> 16) | (tile0 << 16));                 // td1[31:16] | tile_dim0
    g1[4] = (int)(tile1 & 0xffffu);                             // tile_dim1 (tile_dim2=0)
    g1[5] = (int)td0;                                           // tensor_dim0_stride[31:0]
    g1[6] = 0;                                                  // stride[47:32] | dim1_stride
    g1[7] = 0;

    tdm_i4 z4 = {0, 0, 0, 0};                                   // 2D tensor: groups 2/3 unused
    tdm_i8 z8 = {0, 0, 0, 0, 0, 0, 0, 0};
    __builtin_amdgcn_tensor_load_to_lds(g0, g1, z4, z4, z8, /*cpol=*/0);
}

static __device__ __forceinline__ unsigned lds_byte_addr(const void* p) {
    // flat LDS aperture: LDS offset == addr[31:0]
    return (unsigned)(uintptr_t)p;
}

// ---------------------------------------------------------------------------
// 1) fp32 -> bf16 weight conversion (keep (N x K) row-major: each gate-output
//    row of W is a contiguous K-vector == one WMMA B-matrix column).
// ---------------------------------------------------------------------------
__global__ __launch_bounds__(256) void k_cvt_weights(const float* __restrict__ Wih,
                                                     const float* __restrict__ Whh,
                                                     __bf16* __restrict__ WihB,
                                                     __bf16* __restrict__ WhhB) {
    const int n4 = (H3 * H_) / 4;
    int i = blockIdx.x * 256 + threadIdx.x;
    const float* src;
    __bf16* dst;
    int j;
    if (i < n4) { src = Wih; dst = WihB; j = i; }
    else        { j = i - n4; if (j >= n4) return; src = Whh; dst = WhhB; }
    f32x4 v = *(const f32x4*)(src + (size_t)j * 4);
    dst[(size_t)j * 4 + 0] = (__bf16)v.x;
    dst[(size_t)j * 4 + 1] = (__bf16)v.y;
    dst[(size_t)j * 4 + 2] = (__bf16)v.z;
    dst[(size_t)j * 4 + 3] = (__bf16)v.w;
}

// ---------------------------------------------------------------------------
// 2) h0 broadcast init (ping buffer 0)
// ---------------------------------------------------------------------------
__global__ __launch_bounds__(256) void k_init_h(const float* __restrict__ state0,
                                                float* __restrict__ hbuf) {
    int i = blockIdx.x * 256 + threadIdx.x;
    if (i < B_ * H_) hbuf[i] = state0[i & (H_ - 1)];
}

// ---------------------------------------------------------------------------
// 3) expand: xe = tanh(X @ W_e^T + b_e), stored bf16 (BT x H) for the GEMM.
// ---------------------------------------------------------------------------
__global__ __launch_bounds__(256) void k_expand(const float* __restrict__ X,
                                                const float* __restrict__ We,
                                                const float* __restrict__ be,
                                                __bf16* __restrict__ xe) {
    __shared__ float xs[KIN];
    const int bt = blockIdx.x;
    if (threadIdx.x < KIN) xs[threadIdx.x] = X[(size_t)bt * KIN + threadIdx.x];
    __syncthreads();
    for (int h = threadIdx.x; h < H_; h += 256) {
        const float* w = We + (size_t)h * KIN;
        float acc = be[h];
#pragma unroll
        for (int k = 0; k < KIN; ++k) acc += xs[k] * w[k];
        xe[(size_t)bt * H_ + h] = (__bf16)tanhf(acc);
    }
}

// ---------------------------------------------------------------------------
// 4) Big parallel GEMM: xp(BT x 3H, fp32) = xe(BT x H, bf16) @ WihB^T + b_ih.
//    Block tile 128x128, K-chunk 32, 8 waves (2M x 4N), wave tile 64x32.
//    A/B tiles staged by the Tensor Data Mover (wave 0 issues
//    tensor_load_to_lds, tracked with TENSORcnt), double-buffered so the DMA
//    of chunk i+1 overlaps the 8 WMMAs/wave of chunk i.
// ---------------------------------------------------------------------------
__global__ __launch_bounds__(256) void k_xp_gemm(const __bf16* __restrict__ xe,
                                                 const __bf16* __restrict__ Wb,
                                                 const float* __restrict__ bih,
                                                 float* __restrict__ xp) {
    __shared__ __bf16 As[2][128][32];   // 2 x 8 KB
    __shared__ __bf16 Bs[2][128][32];   // 2 x 8 KB
    const int tid  = threadIdx.x;
    const int lane = tid & 31;
    const int wave = tid >> 5;       // 0..7
    const int wm   = wave & 1;       // 2 waves along M
    const int wn   = wave >> 1;      // 4 waves along N
    const int l16  = lane & 15;
    const int g    = lane >> 4;      // lane half-group
    const int m0   = blockIdx.x * 128;   // over BT
    const int n0   = blockIdx.y * 128;   // over 3H

    v8f acc[4][2];
#pragma unroll
    for (int i = 0; i < 4; ++i)
#pragma unroll
        for (int j = 0; j < 2; ++j) acc[i][j] = (v8f){};

    const bool issuer = (wave == 0);           // one wave drives the TDM
    const unsigned ldsA[2] = { lds_byte_addr(&As[0][0][0]), lds_byte_addr(&As[1][0][0]) };
    const unsigned ldsB[2] = { lds_byte_addr(&Bs[0][0][0]), lds_byte_addr(&Bs[1][0][0]) };

    if (issuer) {                               // prime chunk 0 into buffer 0
        tdm_load_tile(ldsA[0], xe + (size_t)m0 * H_ + 0, BT);
        tdm_load_tile(ldsB[0], Wb + (size_t)n0 * H_ + 0, H3);
    }

    const int NCH = H_ / 32;
    for (int ic = 0; ic < NCH; ++ic) {
        const int buf = ic & 1;
        if (issuer) {
            if (ic + 1 < NCH) {                 // prefetch chunk ic+1 into other buffer
                const int kn = (ic + 1) * 32;
                tdm_load_tile(ldsA[buf ^ 1], xe + (size_t)m0 * H_ + kn, BT);
                tdm_load_tile(ldsB[buf ^ 1], Wb + (size_t)n0 * H_ + kn, H3);
                __builtin_amdgcn_s_wait_tensorcnt((short)2);  // in-order TDM: chunk ic landed
            } else {
                __builtin_amdgcn_s_wait_tensorcnt((short)0);
            }
        }
        __syncthreads();                        // release all waves, LDS tile visible

        v16bf afrag[4], bfrag[2];
#pragma unroll
        for (int i = 0; i < 4; ++i) {
            Frag32B f;
            const int row = wm * 64 + i * 16 + l16;
            f.u[0] = *(const u32x4*)(&As[buf][row][g * 8 + 0]);
            f.u[1] = *(const u32x4*)(&As[buf][row][g * 8 + 16]);
            afrag[i] = f.v;
        }
#pragma unroll
        for (int j = 0; j < 2; ++j) {
            Frag32B f;
            const int col = wn * 32 + j * 16 + l16;
            f.u[0] = *(const u32x4*)(&Bs[buf][col][g * 8 + 0]);
            f.u[1] = *(const u32x4*)(&Bs[buf][col][g * 8 + 16]);
            bfrag[j] = f.v;
        }
#pragma unroll
        for (int i = 0; i < 4; ++i)
#pragma unroll
            for (int j = 0; j < 2; ++j)
                acc[i][j] = wmma_bf16(afrag[i], bfrag[j], acc[i][j]);

        __syncthreads();                        // reads done before TDM reuses this buffer
    }

    // Epilogue: C layout = VGPR e, lanes<16 -> M=e, lanes>=16 -> M=8+e; N=lane&15
#pragma unroll
    for (int i = 0; i < 4; ++i)
#pragma unroll
        for (int j = 0; j < 2; ++j) {
            const int col = n0 + wn * 32 + j * 16 + l16;
            const float bias = bih[col];
#pragma unroll
            for (int e = 0; e < 8; ++e) {
                const int row = m0 + wm * 64 + i * 16 + g * 8 + e;
                xp[(size_t)row * H3 + col] = acc[i][j][e] + bias;
            }
        }
}

// ---------------------------------------------------------------------------
// 5) Sequential GRU scan. Grid = 64 WGs = 4 wave-clusters x 16 WGs.
//    Cluster c owns batch rows [16c, 16c+16); WG s within a cluster owns
//    hidden slice [64s, 64s+64). Per step: stage cluster's h (16x1024) into
//    LDS as bf16, compute 16x64 r/z/n gate tiles with WMMA (B fragments
//    streamed straight from L2-resident W_hh bf16), fp32 gate math, write h
//    slice into ping-pong global buffer + states, then cluster barrier
//    (s_barrier_signal/-wait -3); one barrier/step thanks to the ping-pong.
// ---------------------------------------------------------------------------
__global__ __launch_bounds__(128) void k_scan(const float* __restrict__ xp,
                                              const __bf16* __restrict__ WhhB,
                                              const float* __restrict__ bhh,
                                              float* __restrict__ hbuf,
                                              float* __restrict__ states) {
    __shared__ __bf16 hs[16][H_];           // 32 KB: cluster's h, bf16
    const int tid  = threadIdx.x;           // 128 threads = 4 waves
    const int lane = tid & 31;
    const int wave = tid >> 5;              // 0..3
    const int l16  = lane & 15;
    const int g    = lane >> 4;
    const int wg   = blockIdx.x;            // 0..63
    const int b0   = (wg >> 4) * 16;        // batch group base
    const int j0   = (wg & 15) * 64;        // hidden slice base
    const int myj  = j0 + wave * 16;        // this wave's 16 columns

    for (int t = 0; t < T_; ++t) {
        const int cur = t & 1;
        const float* hcur = hbuf + (size_t)cur * (B_ * H_);
        float*       hnxt = hbuf + (size_t)(cur ^ 1) * (B_ * H_);

        // stage h (16 x 1024 fp32 -> bf16 LDS); 4096 float4 / 128 threads
        for (int v = tid; v < 16 * (H_ / 4); v += 128) {
            const int rr = v >> 8, c4 = (v & 255) * 4;
            const f32x4 f = *(const f32x4*)(hcur + (size_t)(b0 + rr) * H_ + c4);
            __bf16* d = &hs[rr][c4];
            d[0] = (__bf16)f.x; d[1] = (__bf16)f.y;
            d[2] = (__bf16)f.z; d[3] = (__bf16)f.w;
        }
        __syncthreads();

        v8f accR = (v8f){}, accZ = (v8f){}, accN = (v8f){};
        for (int kc = 0; kc < H_; kc += 32) {
            Frag32B fa;
            fa.u[0] = *(const u32x4*)(&hs[l16][kc + g * 16 + 0]);
            fa.u[1] = *(const u32x4*)(&hs[l16][kc + g * 16 + 8]);

            const int kb = kc + g * 16;     // 32 contiguous bytes per lane
            const u32x4* pr = (const u32x4*)(WhhB + (size_t)(0    + myj + l16) * H_ + kb);
            const u32x4* pz = (const u32x4*)(WhhB + (size_t)(1024 + myj + l16) * H_ + kb);
            const u32x4* pn = (const u32x4*)(WhhB + (size_t)(2048 + myj + l16) * H_ + kb);
            if (kc + 32 < H_) __builtin_prefetch((const void*)((const char*)pr + 64), 0, 1);
            Frag32B fr, fz, fn;
            fr.u[0] = pr[0]; fr.u[1] = pr[1];
            fz.u[0] = pz[0]; fz.u[1] = pz[1];
            fn.u[0] = pn[0]; fn.u[1] = pn[1];

            accR = wmma_bf16(fa.v, fr.v, accR);
            accZ = wmma_bf16(fa.v, fz.v, accZ);
            accN = wmma_bf16(fa.v, fn.v, accN);
        }

        // gates (fp32): r=sig(xr+hr), z=sig(xz+hz), n=tanh(xn+r*hn)
#pragma unroll
        for (int e = 0; e < 8; ++e) {
            const int b   = b0 + g * 8 + e;   // batch row
            const int col = myj + l16;        // hidden index
            const size_t xb = ((size_t)b * T_ + t) * H3;
            const float hr = accR[e] + bhh[col];
            const float hz = accZ[e] + bhh[1024 + col];
            const float hn = accN[e] + bhh[2048 + col];
            const float rgt = 1.f / (1.f + __expf(-(xp[xb + col]        + hr)));
            const float zgt = 1.f / (1.f + __expf(-(xp[xb + 1024 + col] + hz)));
            const float ngt = tanhf(xp[xb + 2048 + col] + rgt * hn);
            const float hold = hcur[(size_t)b * H_ + col];
            const float hnew = (1.f - zgt) * ngt + zgt * hold;
            hnxt[(size_t)b * H_ + col]            = hnew;
            states[((size_t)b * T_ + t) * H_ + col] = hnew;
        }

        __threadfence();      // make h writes device-visible before cluster sync
        __syncthreads();      // LDS reuse + all waves done
        if (wave == 0) {
            __builtin_amdgcn_s_cluster_barrier();              // signal -3 + wait -3
        } else {
            asm volatile("s_barrier_wait -3" ::: "memory");    // members just wait
        }
    }
}

// ---------------------------------------------------------------------------
// 6) Output projection: out(BT x 32) = states(BT x 1024) @ W_ly^T.
// ---------------------------------------------------------------------------
__global__ __launch_bounds__(256) void k_outproj(const float* __restrict__ states,
                                                 const float* __restrict__ Wly,
                                                 float* __restrict__ out) {
    __shared__ float row[8][H_];     // 32 KB
    const int bt0 = blockIdx.x * 8;
    for (int v = threadIdx.x; v < 8 * (H_ / 4); v += 256) {
        const int rr = v >> 8, c4 = (v & 255) * 4;
        *(f32x4*)(&row[rr][c4]) = *(const f32x4*)(states + (size_t)(bt0 + rr) * H_ + c4);
    }
    __syncthreads();
    const int rr = threadIdx.x >> 5;     // 8 rows x 32 lanes
    const int o  = threadIdx.x & 31;
    const float* w = Wly + (size_t)o * H_;
    float acc = 0.f;
    for (int k = 0; k < H_; ++k) acc += row[rr][k] * w[k];
    out[(size_t)(bt0 + rr) * KOUT + o] = acc;
}

// ---------------------------------------------------------------------------
// Launch. Workspace layout (bytes):
//   xe bf16 (64 MB) | WihB bf16 (6 MB) | WhhB bf16 (6 MB) |
//   hbuf fp32 ping-pong (512 KB) | xp fp32 (384 MB)
// ---------------------------------------------------------------------------
extern "C" void kernel_launch(void* const* d_in, const int* in_sizes, int n_in,
                              void* d_out, int out_size, void* d_ws, size_t ws_size,
                              hipStream_t stream) {
    const float* X      = (const float*)d_in[0];
    const float* state0 = (const float*)d_in[1];
    const float* We     = (const float*)d_in[2];
    const float* be     = (const float*)d_in[3];
    const float* Wih    = (const float*)d_in[4];
    const float* bih    = (const float*)d_in[5];
    const float* Whh    = (const float*)d_in[6];
    const float* bhh    = (const float*)d_in[7];
    const float* Wly    = (const float*)d_in[8];

    float* out    = (float*)d_out;
    float* states = out + (size_t)BT * KOUT;   // outputs then states, concatenated

    char* ws = (char*)d_ws;
    size_t off = 0;
    __bf16* xe   = (__bf16*)(ws + off); off += (size_t)BT * H_ * 2;
    __bf16* WihB = (__bf16*)(ws + off); off += (size_t)H3 * H_ * 2;
    __bf16* WhhB = (__bf16*)(ws + off); off += (size_t)H3 * H_ * 2;
    float*  hbuf = (float*)(ws + off);  off += (size_t)2 * B_ * H_ * 4;
    float*  xp   = (float*)(ws + off);  off += (size_t)BT * H3 * 4;

    k_cvt_weights<<<(2 * (H3 * H_ / 4) + 255) / 256, 256, 0, stream>>>(Wih, Whh, WihB, WhhB);
    k_init_h<<<(B_ * H_ + 255) / 256, 256, 0, stream>>>(state0, hbuf);
    k_expand<<<BT, 256, 0, stream>>>(X, We, be, xe);
    k_xp_gemm<<<dim3(BT / 128, H3 / 128), 256, 0, stream>>>(xe, WihB, bih, xp);
    k_scan<<<64, 128, 0, stream>>>(xp, WhhB, bhh, hbuf, states);
    k_outproj<<<BT / 8, 256, 0, stream>>>(states, Wly, out);
}
